// Linear_Attention_58686433133029
// MI455X (gfx1250) — matrix operand order repaired and checked by reading, not compiled, not run on hardware
//
#include <hip/hip_runtime.h>
#include <hip/hip_bf16.h>
#include <stdint.h>

typedef __bf16 bf16;
typedef __bf16 v16bf __attribute__((ext_vector_type(16)));
typedef float  v8f   __attribute__((ext_vector_type(8)));

#define EPS_  1e-5f
#define B_    4
#define L_    4096
#define E_    1024
#define H_    16
#define D_    64
#define NTOK  (B_*L_)      // 16384

#if defined(__has_builtin)
#if __has_builtin(__builtin_amdgcn_tensor_load_to_lds) && \
    __has_builtin(__builtin_amdgcn_s_wait_tensorcnt)
#define HAVE_TDM 1
#endif
#endif

// ---------------------------------------------------------------- WMMA core
__device__ __forceinline__ v8f wmma_bf16(v16bf a, v16bf b, v8f c) {
  // (neg_a, A, neg_b, B, c_mod, C, reuse_a, reuse_b)
  return __builtin_amdgcn_wmma_f32_16x16x32_bf16(false, a, false, b,
                                                 (short)0, c, false, false);
}

// A operand 16x32 (MxK), source stored row-major src[m][k], leading dim ld.
__device__ __forceinline__ v16bf frag_a_row(const bf16* base, int ld, int lane) {
  const bf16* p = base + (size_t)(lane & 15) * ld + ((lane >> 4) << 3);
  v16bf f;
#pragma unroll
  for (int e = 0; e < 8; ++e) f[e] = p[e];
#pragma unroll
  for (int e = 0; e < 8; ++e) f[e + 8] = p[16 + e];
  return f;
}

// A operand 16x32 where A[m][k] = src[k][m] (transposed source), leading dim ld.
__device__ __forceinline__ v16bf frag_a_col(const bf16* base, int ld, int lane) {
  const int m = lane & 15;
  v16bf f;
#pragma unroll
  for (int e = 0; e < 16; ++e) {
    const int k = (e & 7) + ((lane >> 4) << 3) + ((e >> 3) << 4);
    f[e] = base[(size_t)k * ld + m];
  }
  return f;
}

// B operand 32x16 (KxN) where B[k][n] = src[n][k] (contiguous-K source).
__device__ __forceinline__ v16bf frag_b_col(const bf16* base, int ld, int lane) {
  const bf16* p = base + (size_t)(lane & 15) * ld + ((lane >> 4) << 3);
  v16bf f;
#pragma unroll
  for (int e = 0; e < 16; ++e) f[e] = p[e + ((e >> 3) << 3)];
  return f;
}

// B operand 32x16 where B[k][n] = src[k][n] (row-major KxN source).
__device__ __forceinline__ v16bf frag_b_row(const bf16* base, int ld, int lane) {
  const int n  = lane & 15;
  const int k0 = (lane >> 4) << 3;
  v16bf f;
#pragma unroll
  for (int e = 0; e < 16; ++e) f[e] = base[(size_t)(k0 + e + ((e >> 3) << 3)) * ld + n];
  return f;
}

// ----------------------------------------------------------- TDM 2D tile load
#ifdef HAVE_TDM
typedef unsigned int v4u __attribute__((ext_vector_type(4)));
typedef int          v8i __attribute__((ext_vector_type(8)));
typedef int          v4i __attribute__((ext_vector_type(4)));

// Load a 2D tile (rows x row_units*4 bytes) global->LDS with optional per-row
// LDS padding (pad every 2^(pad_int_code+1) DWORDs by pad_amt_code+1 DWORDs).
// data_size = 4B units.  Descriptor per CDNA5 ISA ch.8 (D# groups 0/1).
__device__ __forceinline__ void tdm_load_2d(unsigned lds_off,
                                            const void* gptr,
                                            unsigned rows,
                                            unsigned row_units,
                                            unsigned pad_int_code,
                                            unsigned pad_amt_code,
                                            unsigned pad_en) {
  const unsigned long long ga = (unsigned long long)(uintptr_t)gptr;
  v4u g0;
  g0[0] = 1u;                                            // count=1, user D#
  g0[1] = lds_off;                                       // lds_addr
  g0[2] = (unsigned)(ga & 0xffffffffu);                  // global_addr[31:0]
  g0[3] = (unsigned)((ga >> 32) & 0x1ffffffu) | (2u << 30); // addr[56:32]|type=2
  v8i g1;
  g1[0] = (int)((2u << 16) | (pad_en << 20) |
                (pad_int_code << 22) | (pad_amt_code << 25)); // 4B elems + pad
  g1[1] = (int)((row_units & 0xffffu) << 16);            // tensor_dim0[15:0]
  g1[2] = (int)(((row_units >> 16) & 0xffffu) |
                ((rows & 0xffffu) << 16));               // dim0[31:16]|dim1[15:0]
  g1[3] = (int)(((rows >> 16) & 0xffffu) |
                ((row_units & 0xffffu) << 16));          // dim1[31:16]|tile_dim0
  g1[4] = (int)(rows & 0xffffu);                         // tile_dim1 (tile_dim2=0)
  g1[5] = (int)row_units;                                // tensor_dim0_stride
  g1[6] = 0;
  g1[7] = 0;
  const v4i z4 = {0, 0, 0, 0};                           // groups 2/3 unused (2D)
  const v8i z8 = {0, 0, 0, 0, 0, 0, 0, 0};
  // 6-arg toolchain flavor: (g0, g1, g2, g3, g4, cpol)
  __builtin_amdgcn_tensor_load_to_lds(g0, g1, z4, z4, z8, 0);
}
#endif

// ------------------------------------------------------------- QKV GEMM(+phi)
#define BM 128
#define BN 128
#define BK 32
#define LDA 40   // padded row (bf16): 80 bytes, 16B-aligned rows

__global__ __launch_bounds__(256)
void qkv_gemm_phi(const float* __restrict__ X, const float* __restrict__ W,
                  bf16* __restrict__ out_bhld, int act) {
  __shared__ alignas(16) bf16 As[BM][LDA];
  __shared__ alignas(16) bf16 Bs[BN][LDA];
  const int m0 = blockIdx.x * BM, n0 = blockIdx.y * BN;
  const int tid = threadIdx.x, lane = tid & 31, w = tid >> 5;
  const int wm = w & 1, wn = w >> 1;   // 2x4 wave grid: 64-row x 32-col slabs

  v8f acc[4][2];
#pragma unroll
  for (int i = 0; i < 4; ++i)
#pragma unroll
    for (int j = 0; j < 2; ++j)
#pragma unroll
      for (int r = 0; r < 8; ++r) acc[i][j][r] = 0.0f;

  for (int kb = 0; kb < E_; kb += BK) {
    if (kb + BK < E_) {   // prefetch next K-tile (global_prefetch_b8)
      const int r = tid >> 3, c = (tid & 7) << 2;
      __builtin_prefetch(X + (size_t)(m0 + r) * E_ + kb + BK + c, 0, 0);
      __builtin_prefetch(W + (size_t)(n0 + r) * E_ + kb + BK + c, 0, 0);
    }
#pragma unroll
    for (int i = 0; i < 4; ++i) {                 // 1024 float4 slots / 256 thr
      const int e = tid + i * 256;
      const int r = e >> 3, c = (e & 7) << 2;
      const float4 va = *reinterpret_cast<const float4*>(
          X + (size_t)(m0 + r) * E_ + kb + c);
      As[r][c + 0] = (bf16)va.x; As[r][c + 1] = (bf16)va.y;
      As[r][c + 2] = (bf16)va.z; As[r][c + 3] = (bf16)va.w;
      const float4 vb = *reinterpret_cast<const float4*>(
          W + (size_t)(n0 + r) * E_ + kb + c);
      Bs[r][c + 0] = (bf16)vb.x; Bs[r][c + 1] = (bf16)vb.y;
      Bs[r][c + 2] = (bf16)vb.z; Bs[r][c + 3] = (bf16)vb.w;
    }
    __syncthreads();
    v16bf af[4], bfr[2];
#pragma unroll
    for (int i = 0; i < 4; ++i) af[i]  = frag_a_row(&As[wm * 64 + i * 16][0], LDA, lane);
#pragma unroll
    for (int j = 0; j < 2; ++j) bfr[j] = frag_b_col(&Bs[wn * 32 + j * 16][0], LDA, lane);
#pragma unroll
    for (int i = 0; i < 4; ++i)
#pragma unroll
      for (int j = 0; j < 2; ++j) acc[i][j] = wmma_bf16(af[i], bfr[j], acc[i][j]);
    __syncthreads();
  }

#pragma unroll
  for (int i = 0; i < 4; ++i)
#pragma unroll
    for (int j = 0; j < 2; ++j) {
      const int col = n0 + wn * 32 + j * 16 + (lane & 15);
      const int h = col >> 6, d = col & (D_ - 1);
#pragma unroll
      for (int r = 0; r < 8; ++r) {
        const int row = m0 + wm * 64 + i * 16 + ((lane >> 4) << 3) + r;
        const int b = row >> 12, t = row & (L_ - 1);
        float v = acc[i][j][r];
        if (act) v = (v > 0.0f) ? (v + 1.0f) : __expf(v);   // elu(x)+1
        out_bhld[(((size_t)(b * H_ + h)) * L_ + t) * D_ + d] = (bf16)v;
      }
    }
}

// -------------------------------------------------- chunked causal linear scan
#define CHK 64
#define ALD 72   // padded row: 144 bytes (128B data + 16B pad), 16B-aligned

__global__ __launch_bounds__(256)
void attn_scan(const bf16* __restrict__ phiQ, const bf16* __restrict__ phiK,
               const bf16* __restrict__ Vv, bf16* __restrict__ attn) {
  __shared__ alignas(16) bf16 Qc[CHK][ALD];
  __shared__ alignas(16) bf16 Kc[CHK][ALD];
  __shared__ alignas(16) bf16 Vc[CHK][ALD];
  __shared__ alignas(16) bf16 Am[CHK][ALD];   // masked intra scores (bf16)
  __shared__ alignas(16) bf16 Sb[D_][ALD];    // bf16 shadow of state S
  __shared__ float Sf[D_][D_];                // f32 master state
  __shared__ float zsh[D_];
  __shared__ float den[CHK];

  const int bh = blockIdx.x;                  // 64 = B*H workgroups
  const int b = bh >> 4, h = bh & 15;
  const int tid = threadIdx.x, lane = tid & 31, w = tid >> 5;

  for (int i = tid; i < D_ * D_; i += 256) {
    Sf[i / D_][i % D_] = 0.0f;
    Sb[i / D_][i % D_] = (bf16)0.0f;
  }
  if (tid < D_) zsh[tid] = 0.0f;

  const bf16* qb = phiQ + (size_t)bh * L_ * D_;
  const bf16* kb = phiK + (size_t)bh * L_ * D_;
  const bf16* vb = Vv   + (size_t)bh * L_ * D_;
  bf16* ob = attn + (size_t)b * L_ * E_ + (size_t)h * D_;   // row stride E_
  __syncthreads();

  for (int c0 = 0; c0 < L_; c0 += CHK) {
    // -------- stage chunk tiles (64 rows x 128B each) global -> LDS
#ifdef HAVE_TDM
    // TDM: one wave issues three D#-driven DMA loads; padding of 4 DWORDs
    // (16B) every 32 DWORDs (128B row) produces the ALD=72 layout in-flight.
    if (w == 0) {
      const unsigned roww = D_ / 2;             // 32 x 4B units per row
      tdm_load_2d((unsigned)(uintptr_t)&Qc[0][0], qb + (size_t)c0 * D_,
                  CHK, roww, 4u, 3u, 1u);
      tdm_load_2d((unsigned)(uintptr_t)&Kc[0][0], kb + (size_t)c0 * D_,
                  CHK, roww, 4u, 3u, 1u);
      tdm_load_2d((unsigned)(uintptr_t)&Vc[0][0], vb + (size_t)c0 * D_,
                  CHK, roww, 4u, 3u, 1u);
      __builtin_amdgcn_s_wait_tensorcnt(0);
    }
#else
    for (int i = tid; i < (CHK * D_) / 8; i += 256) {
      const int r = i >> 3, c = (i & 7) << 3;
      const size_t g = (size_t)(c0 + r) * D_ + c;
      *reinterpret_cast<uint4*>(&Qc[r][c]) = *reinterpret_cast<const uint4*>(qb + g);
      *reinterpret_cast<uint4*>(&Kc[r][c]) = *reinterpret_cast<const uint4*>(kb + g);
      *reinterpret_cast<uint4*>(&Vc[r][c]) = *reinterpret_cast<const uint4*>(vb + g);
    }
#endif
    __syncthreads();

    // -------- A_intra = phiQ . phiK^T, causal mask, stash as bf16
#pragma unroll
    for (int tt = 0; tt < 2; ++tt) {
      const int ti = w * 2 + tt, tm = ti >> 2, tn = ti & 3;
      v8f acc;
#pragma unroll
      for (int r = 0; r < 8; ++r) acc[r] = 0.0f;
#pragma unroll
      for (int k2 = 0; k2 < D_; k2 += 32)
        acc = wmma_bf16(frag_a_row(&Qc[tm * 16][k2], ALD, lane),
                        frag_b_col(&Kc[tn * 16][k2], ALD, lane), acc);
      const int cj = tn * 16 + (lane & 15);
#pragma unroll
      for (int r = 0; r < 8; ++r) {
        const int ri = tm * 16 + ((lane >> 4) << 3) + r;
        Am[ri][cj] = (bf16)((ri >= cj) ? acc[r] : 0.0f);
      }
    }
    __syncthreads();

    // -------- den_t = rowsum(masked A) + phiq_t . z_prev + eps
    if (tid < CHK) {
      float s = EPS_;
#pragma unroll 8
      for (int d = 0; d < D_; ++d) s += (float)Qc[tid][d] * zsh[d];
      for (int j = 0; j <= tid; ++j) s += (float)Am[tid][j];
      den[tid] = s;
    }
    __syncthreads();

    // -------- num = Am.V + phiQ.S_prev ; out = num / den
#pragma unroll
    for (int tt = 0; tt < 2; ++tt) {
      const int ti = w * 2 + tt, tm = ti >> 2, tn = ti & 3;
      v8f acc;
#pragma unroll
      for (int r = 0; r < 8; ++r) acc[r] = 0.0f;
#pragma unroll
      for (int k2 = 0; k2 < CHK; k2 += 32)
        acc = wmma_bf16(frag_a_row(&Am[tm * 16][k2], ALD, lane),
                        frag_b_row(&Vc[k2][tn * 16], ALD, lane), acc);
#pragma unroll
      for (int k2 = 0; k2 < D_; k2 += 32)
        acc = wmma_bf16(frag_a_row(&Qc[tm * 16][k2], ALD, lane),
                        frag_b_row(&Sb[k2][tn * 16], ALD, lane), acc);
      const int col = tn * 16 + (lane & 15);
#pragma unroll
      for (int r = 0; r < 8; ++r) {
        const int row = tm * 16 + ((lane >> 4) << 3) + r;
        ob[(size_t)(c0 + row) * E_ + col] = (bf16)(acc[r] / den[row]);
      }
    }
    __syncthreads();   // Sb consumed above; safe to update below

    // -------- S += phiK^T . V ; z += colsum(phiK)
#pragma unroll
    for (int tt = 0; tt < 2; ++tt) {
      const int ti = w * 2 + tt, tm = ti >> 2, tn = ti & 3;
      v8f acc;
#pragma unroll
      for (int r = 0; r < 8; ++r) acc[r] = 0.0f;
#pragma unroll
      for (int k2 = 0; k2 < CHK; k2 += 32)
        acc = wmma_bf16(frag_a_col(&Kc[k2][tm * 16], ALD, lane),
                        frag_b_row(&Vc[k2][tn * 16], ALD, lane), acc);
      const int ec = tn * 16 + (lane & 15);
#pragma unroll
      for (int r = 0; r < 8; ++r) {
        const int dd = tm * 16 + ((lane >> 4) << 3) + r;
        const float ns = Sf[dd][ec] + acc[r];
        Sf[dd][ec] = ns;
        Sb[dd][ec] = (bf16)ns;
      }
    }
    if (tid < D_) {
      float s = zsh[tid];
#pragma unroll 8
      for (int t2 = 0; t2 < CHK; ++t2) s += (float)Kc[t2][tid];
      zsh[tid] = s;
    }
    __syncthreads();
  }
}

// ----------------------------------------------------- output-projection GEMM
__global__ __launch_bounds__(256)
void out_proj_gemm(const bf16* __restrict__ A, const float* __restrict__ W,
                   const float* __restrict__ bias, float* __restrict__ Y) {
  __shared__ alignas(16) bf16 As[BM][LDA];
  __shared__ alignas(16) bf16 Bs[BN][LDA];
  const int m0 = blockIdx.x * BM, n0 = blockIdx.y * BN;
  const int tid = threadIdx.x, lane = tid & 31, w = tid >> 5;
  const int wm = w & 1, wn = w >> 1;

  v8f acc[4][2];
#pragma unroll
  for (int i = 0; i < 4; ++i)
#pragma unroll
    for (int j = 0; j < 2; ++j)
#pragma unroll
      for (int r = 0; r < 8; ++r) acc[i][j][r] = 0.0f;

  for (int kbb = 0; kbb < E_; kbb += BK) {
    if (kbb + BK < E_) {
      const int r = tid >> 3, c = (tid & 7) << 2;
      __builtin_prefetch(A + (size_t)(m0 + (tid >> 2)) * E_ + kbb + BK, 0, 0);
      __builtin_prefetch(W + (size_t)(n0 + r) * E_ + kbb + BK + c, 0, 0);
    }
#pragma unroll
    for (int i = 0; i < 2; ++i) {                 // A tile: 512 uint4 slots
      const int e = tid + i * 256;
      const int r = e >> 2, c = (e & 3) << 3;
      *reinterpret_cast<uint4*>(&As[r][c]) =
          *reinterpret_cast<const uint4*>(A + (size_t)(m0 + r) * E_ + kbb + c);
    }
#pragma unroll
    for (int i = 0; i < 4; ++i) {                 // W tile: f32 -> bf16
      const int e = tid + i * 256;
      const int r = e >> 3, c = (e & 7) << 2;
      const float4 vb = *reinterpret_cast<const float4*>(
          W + (size_t)(n0 + r) * E_ + kbb + c);
      Bs[r][c + 0] = (bf16)vb.x; Bs[r][c + 1] = (bf16)vb.y;
      Bs[r][c + 2] = (bf16)vb.z; Bs[r][c + 3] = (bf16)vb.w;
    }
    __syncthreads();
    v16bf af[4], bfr[2];
#pragma unroll
    for (int i = 0; i < 4; ++i) af[i]  = frag_a_row(&As[wm * 64 + i * 16][0], LDA, lane);
#pragma unroll
    for (int j = 0; j < 2; ++j) bfr[j] = frag_b_col(&Bs[wn * 32 + j * 16][0], LDA, lane);
#pragma unroll
    for (int i = 0; i < 4; ++i)
#pragma unroll
      for (int j = 0; j < 2; ++j) acc[i][j] = wmma_bf16(af[i], bfr[j], acc[i][j]);
    __syncthreads();
  }

#pragma unroll
  for (int i = 0; i < 4; ++i)
#pragma unroll
    for (int j = 0; j < 2; ++j) {
      const int col = n0 + wn * 32 + j * 16 + (lane & 15);
      const float bv = bias[col];
#pragma unroll
      for (int r = 0; r < 8; ++r) {
        const int row = m0 + wm * 64 + i * 16 + ((lane >> 4) << 3) + r;
        Y[(size_t)row * E_ + col] = acc[i][j][r] + bv;
      }
    }
}

// --------------------------------------------------------------------- launch
extern "C" void kernel_launch(void* const* d_in, const int* in_sizes, int n_in,
                              void* d_out, int out_size, void* d_ws, size_t ws_size,
                              hipStream_t stream) {
  const float* x  = (const float*)d_in[0];
  const float* Wq = (const float*)d_in[1];
  const float* Wk = (const float*)d_in[2];
  const float* Wv = (const float*)d_in[3];
  const float* Wo = (const float*)d_in[4];
  const float* bo = (const float*)d_in[5];

  char* ws = (char*)d_ws;
  const size_t mat = (size_t)NTOK * E_ * sizeof(bf16);   // 32 MB each
  bf16* phiQ = (bf16*)(ws + 0 * mat);   // [B,H,L,D]
  bf16* phiK = (bf16*)(ws + 1 * mat);   // [B,H,L,D]
  bf16* Vb   = (bf16*)(ws + 2 * mat);   // [B,H,L,D]
  bf16* Attn = (bf16*)(ws + 3 * mat);   // [B*L, E] row-major

  const dim3 g(NTOK / BM, E_ / BN), blk(256);
  qkv_gemm_phi<<<g, blk, 0, stream>>>(x, Wq, phiQ, 1);
  qkv_gemm_phi<<<g, blk, 0, stream>>>(x, Wk, phiK, 1);
  qkv_gemm_phi<<<g, blk, 0, stream>>>(x, Wv, Vb, 0);
  attn_scan<<<dim3(B_ * H_), blk, 0, stream>>>(phiQ, phiK, Vb, Attn);
  out_proj_gemm<<<g, blk, 0, stream>>>(Attn, Wo, bo, (float*)d_out);
}
